// RVM_Block_37589553775311
// MI455X (gfx1250) — matrix operand order repaired and checked by reading, not compiled, not run on hardware
//
#include <hip/hip_runtime.h>

// ---------------------------------------------------------------------------
// Sizes (from reference)
// ---------------------------------------------------------------------------
#define BATCH     8
#define C_IN      256
#define NPOS      4096          // 64*64
#define MB        32            // 4 chunks * 8 batch  (mamba batch)
#define DMODEL    64
#define DINNER    128
#define DSTATE    16
#define DTRANK    4
#define DCONV     4

typedef __attribute__((ext_vector_type(16))) __bf16 v16bf;
typedef __attribute__((ext_vector_type(8)))  float  v8f;
typedef __attribute__((ext_vector_type(4)))  unsigned int u32x4;

union Frag16 { v16bf v; u32x4 q[2]; };

// ---------------------------------------------------------------------------
// bf16 helpers (RNE)
// ---------------------------------------------------------------------------
__device__ __forceinline__ unsigned short f2bfbits(float f) {
    union { float f; unsigned u; } uf; uf.f = f;
    return (unsigned short)((uf.u + 0x7FFFu + ((uf.u >> 16) & 1u)) >> 16);
}
__device__ __forceinline__ float silu_f(float x) {
    return x / (1.f + __expf(-x));
}

// ---------------------------------------------------------------------------
// Kernel P: one-shot weight conversion to bf16
// layout in dst: [0,16384) in_proj_w | [16384,24576) out_proj_w | [24576,90112) proj_w
// ---------------------------------------------------------------------------
__global__ void k_prep_w(const float* __restrict__ w0, const float* __restrict__ w1,
                         const float* __restrict__ w2, unsigned short* __restrict__ dst) {
    const int i = blockIdx.x * 256 + threadIdx.x;
    if (i < 16384)       dst[i] = f2bfbits(w0[i]);
    else if (i < 24576)  dst[i] = f2bfbits(w1[i - 16384]);
    else if (i < 90112)  dst[i] = f2bfbits(w2[i - 24576]);
}

// ---------------------------------------------------------------------------
// Kernel 0: transpose x (B, 256, 4096) -> xflat (B, 4096, 256)
// ---------------------------------------------------------------------------
__global__ void k_transpose(const float* __restrict__ x, float* __restrict__ xflat) {
    __shared__ float tile[32][33];
    const int b  = blockIdx.z;
    const int i0 = blockIdx.x * 32;
    const int c0 = blockIdx.y * 32;
    const int tx = threadIdx.x, ty = threadIdx.y; // (32, 8)
    const float* src = x     + (long)b * C_IN * NPOS;
    float*       dst = xflat + (long)b * NPOS * C_IN;
#pragma unroll
    for (int k = 0; k < 32; k += 8)
        tile[ty + k][tx] = src[(long)(c0 + ty + k) * NPOS + i0 + tx];
    __syncthreads();
#pragma unroll
    for (int k = 0; k < 32; k += 8)
        dst[(long)(i0 + ty + k) * C_IN + c0 + tx] = tile[tx][ty + k];
}

// ---------------------------------------------------------------------------
// Kernel 1: LayerNorm rows of xflat -> chunks fp32 (skip path) + bf16 (GEMM A)
// ---------------------------------------------------------------------------
__global__ void k_ln1(const float* __restrict__ xflat,
                      const float* __restrict__ nw, const float* __restrict__ nb,
                      float* __restrict__ chunks, unsigned short* __restrict__ chunksbf) {
    const int b = blockIdx.x >> 12;
    const int i = blockIdx.x & 4095;
    const int c = threadIdx.x; // 0..255
    __shared__ float red[256];
    const float v = xflat[((long)b * NPOS + i) * C_IN + c];
    red[c] = v; __syncthreads();
    for (int s = 128; s > 0; s >>= 1) { if (c < s) red[c] += red[c + s]; __syncthreads(); }
    const float mean = red[0] * (1.f / 256.f);
    __syncthreads();
    const float dv = v - mean;
    red[c] = dv * dv; __syncthreads();
    for (int s = 128; s > 0; s >>= 1) { if (c < s) red[c] += red[c + s]; __syncthreads(); }
    const float var = red[0] * (1.f / 256.f);
    const float o = dv * rsqrtf(var + 1e-5f) * nw[c] + nb[c];
    const int chunk = c >> 6, d = c & 63;
    const long oi = ((long)(chunk * BATCH + b) * NPOS + i) * DMODEL + d;
    chunks[oi]   = o;
    chunksbf[oi] = f2bfbits(o);
}

// ---------------------------------------------------------------------------
// WMMA GEMM (bf16 x bf16 -> fp32): C[z] (MxN) = A[z] (MxK) * W^T, W (NxK).
// One wave = one 16x16 tile; 8 waves/block tile M; b128 fragment loads.
// ---------------------------------------------------------------------------
__global__ void k_gemm_bf16(const unsigned short* __restrict__ A,
                            const unsigned short* __restrict__ W,
                            float* __restrict__ C, int K, int lda, int ldw, int ldc,
                            long abatch, long cbatch) {
    const int lane = threadIdx.x & 31;
    const int wv   = threadIdx.x >> 5;
    const int m0   = (blockIdx.x * 8 + wv) * 16;
    const int n0   = blockIdx.y * 16;
    const long z   = blockIdx.z;
    A += z * abatch; C += z * cbatch;
    const int r = lane & 15, hf = lane >> 4;
    const unsigned short* abase = A + (long)(m0 + r) * lda + hf * 8;
    const unsigned short* wbase = W + (long)(n0 + r) * ldw + hf * 8;
    v8f acc = {};
    for (int k0 = 0; k0 < K; k0 += 32) {
        Frag16 a, bf;
        a.q[0]  = *(const u32x4*)(abase + k0);
        a.q[1]  = *(const u32x4*)(abase + k0 + 16);
        bf.q[0] = *(const u32x4*)(wbase + k0);
        bf.q[1] = *(const u32x4*)(wbase + k0 + 16);
        acc = __builtin_amdgcn_wmma_f32_16x16x32_bf16(false, a.v, false, bf.v,
                                                      (short)0, acc, false, false);
    }
    float* crow = C + (long)(m0 + hf * 8) * ldc + n0 + r;
#pragma unroll
    for (int v = 0; v < 8; ++v) crow[(long)v * ldc] = acc[v];
}

// ---------------------------------------------------------------------------
// Kernel 3: causal depthwise conv1d (k=4) + bias + SiLU on xi half of xz
// ---------------------------------------------------------------------------
__global__ void k_conv_silu(const float* __restrict__ xz, const float* __restrict__ cw,
                            const float* __restrict__ cb, float* __restrict__ xi) {
    const long idx = (long)blockIdx.x * 256 + threadIdx.x; // over 32*4096*128
    const int  d  = (int)(idx & 127);
    const long bl = idx >> 7;            // mb*4096 + l
    const int  l  = (int)(bl & 4095);
    float acc = cb[d];
#pragma unroll
    for (int t = 0; t < DCONV; ++t) {
        const int ls = l - (DCONV - 1) + t;
        if (ls >= 0) acc += cw[d * DCONV + t] * xz[(bl - (DCONV - 1) + t) * 256 + d];
    }
    xi[idx] = acc * (1.f / (1.f + __expf(-acc)));
}

// ---------------------------------------------------------------------------
// Kernel 4: fused x_proj (128->36) + dt_proj (4->128) + softplus
// ---------------------------------------------------------------------------
__global__ void k_xproj_dt(const float* __restrict__ xi, const float* __restrict__ xw,
                           const float* __restrict__ dtw, const float* __restrict__ dtb,
                           float* __restrict__ dt, float* __restrict__ Bm,
                           float* __restrict__ Cm) {
    const long row = blockIdx.x;         // mb*4096 + l
    const int  t   = threadIdx.x;        // 0..127
    __shared__ float rbuf[DINNER];
    __shared__ float xd[DTRANK + 2 * DSTATE];
    rbuf[t] = xi[row * DINNER + t];
    __syncthreads();
    if (t < DTRANK + 2 * DSTATE) {
        float acc = 0.f;
        const float* w = xw + t * DINNER;
#pragma unroll 8
        for (int k = 0; k < DINNER; ++k) acc += rbuf[k] * w[k];
        xd[t] = acc;
    }
    __syncthreads();
    if (t < DSTATE)          Bm[row * DSTATE + t]            = xd[DTRANK + t];
    else if (t < 2 * DSTATE) Cm[row * DSTATE + (t - DSTATE)] = xd[DTRANK + DSTATE + (t - DSTATE)];
    float a = dtb[t];
#pragma unroll
    for (int rr = 0; rr < DTRANK; ++rr) a += xd[rr] * dtw[t * DTRANK + rr];
    dt[row * DINNER + t] = (a > 20.f) ? a : __logf(1.f + __expf(a));
}

// ---------------------------------------------------------------------------
// Kernel 5: selective scan. One block per mamba batch, one thread per channel.
// B/C tiles in LDS (2 barriers / 128 steps); prefetch next dt/xi tile.
// ---------------------------------------------------------------------------
__global__ void k_scan(const float* __restrict__ dt, const float* __restrict__ Bm,
                       const float* __restrict__ Cm, const float* __restrict__ xi,
                       const float* __restrict__ A_log, float* __restrict__ ys) {
    const int b = blockIdx.x;   // 0..31
    const int d = threadIdx.x;  // 0..127
    __shared__ float Bs[128 * DSTATE];
    __shared__ float Cs[128 * DSTATE];
    float A[DSTATE], h[DSTATE];
#pragma unroll
    for (int n = 0; n < DSTATE; ++n) { A[n] = -__expf(A_log[d * DSTATE + n]); h[n] = 0.f; }
    const long base = (long)b * NPOS;
    for (int l0 = 0; l0 < NPOS; l0 += 128) {
        if (l0 + 128 < NPOS) {  // gfx1250 global_prefetch of the next streaming tile
            __builtin_prefetch(&dt[(base + l0 + 128) * DINNER + d], 0, 1);
            __builtin_prefetch(&xi[(base + l0 + 128) * DINNER + d], 0, 1);
        }
        for (int idx = d; idx < 128 * DSTATE; idx += 128) {
            Bs[idx] = Bm[(base + l0) * DSTATE + idx];
            Cs[idx] = Cm[(base + l0) * DSTATE + idx];
        }
        __syncthreads();
        for (int j = 0; j < 128; ++j) {
            const long li  = base + l0 + j;
            const float dv = dt[li * DINNER + d];
            const float u  = xi[li * DINNER + d];
            const float du = dv * u;
            float accv = 0.f;
#pragma unroll
            for (int n = 0; n < DSTATE; ++n) {
                const float dA = __expf(dv * A[n]);
                h[n] = dA * h[n] + du * Bs[j * DSTATE + n];
                accv += h[n] * Cs[j * DSTATE + n];
            }
            ys[li * DINNER + d] = accv;
        }
        __syncthreads();
    }
}

// ---------------------------------------------------------------------------
// Kernel 6: gate: g = (ys + D*xi) * silu(z)  -> bf16 buffer for out_proj
// ---------------------------------------------------------------------------
__global__ void k_gate(const float* __restrict__ ys, const float* __restrict__ xi,
                       const float* __restrict__ xz, const float* __restrict__ Dp,
                       unsigned short* __restrict__ g) {
    const long idx = (long)blockIdx.x * 256 + threadIdx.x; // over 32*4096*128
    const int  d   = (int)(idx & 127);
    const long bl  = idx >> 7;
    const float z  = xz[bl * 256 + 128 + d];
    const float v  = (ys[idx] + Dp[d] * xi[idx]) * silu_f(z);
    g[idx] = f2bfbits(v);
}

// ---------------------------------------------------------------------------
// Kernel 8: rotary(sequence-dim rotate) + skip + recombine + LN -> xm bf16
// ---------------------------------------------------------------------------
__global__ void k_rot_skip_ln(const float* __restrict__ yo, const float* __restrict__ chunks,
                              const float* __restrict__ xflat, const float* __restrict__ nw,
                              const float* __restrict__ nb, const float* __restrict__ ssp,
                              unsigned short* __restrict__ xm) {
    const int b = blockIdx.x >> 12;
    const int i = blockIdx.x & 4095;
    const int c = threadIdx.x;                  // 0..255
    const int chunk = c >> 6, d = c & 63;
    const int mb = chunk * BATCH + b;
    const float ss = ssp[0];
    const long ybase = (long)mb * NPOS * DMODEL;
    const float yv = yo[ybase + (long)i * DMODEL + d];
    int i2; float sgn;
    if (i < NPOS / 2) { i2 = 2 * i + 1; sgn = -1.f; }
    else              { i2 = 2 * (i - NPOS / 2); sgn = 1.f; }
    const float rv = sgn * yo[ybase + (long)i2 * DMODEL + d];
    const float ex  = (float)(d >> 1) * (1.f / 31.f);
    const float ang = __expf(-ex * 6.907755278982137f); // ln(1000)
    const float ph  = (float)i * ang;
    float s, cv; __sincosf(ph, &s, &cv);
    const float ts   = yv * cv + rv * s;
    const float comb = ts + ss * chunks[ybase + (long)i * DMODEL + d];
    const float t    = comb + ss * xflat[((long)b * NPOS + i) * C_IN + c];
    __shared__ float red[256];
    red[c] = t; __syncthreads();
    for (int s2 = 128; s2 > 0; s2 >>= 1) { if (c < s2) red[c] += red[c + s2]; __syncthreads(); }
    const float mean = red[0] * (1.f / 256.f);
    __syncthreads();
    const float dvv = t - mean;
    red[c] = dvv * dvv; __syncthreads();
    for (int s2 = 128; s2 > 0; s2 >>= 1) { if (c < s2) red[c] += red[c + s2]; __syncthreads(); }
    const float var = red[0] * (1.f / 256.f);
    const float o = dvv * rsqrtf(var + 1e-5f) * nw[c] + nb[c];
    xm[((long)b * NPOS + i) * C_IN + c] = f2bfbits(o);
}

// ---------------------------------------------------------------------------
// Kernel 9: final proj WMMA (32768 x 256 x 256) + bias, transposed store
// ---------------------------------------------------------------------------
__global__ void k_gemm_final(const unsigned short* __restrict__ A,
                             const unsigned short* __restrict__ W,
                             const float* __restrict__ bias, float* __restrict__ out) {
    const int lane = threadIdx.x & 31;
    const int wv   = threadIdx.x >> 5;
    const int m0   = (blockIdx.x * 8 + wv) * 16;
    const int n0   = blockIdx.y * 16;
    const int r = lane & 15, hf = lane >> 4;
    const unsigned short* abase = A + (long)(m0 + r) * C_IN + hf * 8;
    const unsigned short* wbase = W + (long)(n0 + r) * C_IN + hf * 8;
    v8f acc = {};
#pragma unroll
    for (int k0 = 0; k0 < C_IN; k0 += 32) {
        Frag16 a, bf;
        a.q[0]  = *(const u32x4*)(abase + k0);
        a.q[1]  = *(const u32x4*)(abase + k0 + 16);
        bf.q[0] = *(const u32x4*)(wbase + k0);
        bf.q[1] = *(const u32x4*)(wbase + k0 + 16);
        acc = __builtin_amdgcn_wmma_f32_16x16x32_bf16(false, a.v, false, bf.v,
                                                      (short)0, acc, false, false);
    }
    const int n  = n0 + r;
    const float bv = bias[n];
    const int mrow = m0 + hf * 8;
#pragma unroll
    for (int v = 0; v < 8; ++v) {
        const int m = mrow + v;
        const int b = m >> 12;       // /4096
        const int i = m & 4095;
        out[((long)b * C_IN + n) * NPOS + i] = acc[v] + bv;
    }
}

// ---------------------------------------------------------------------------
// Host launcher
// ---------------------------------------------------------------------------
extern "C" void kernel_launch(void* const* d_in, const int* in_sizes, int n_in,
                              void* d_out, int out_size, void* d_ws, size_t ws_size,
                              hipStream_t stream) {
    const float* x        = (const float*)d_in[0];
    const float* norm_w   = (const float*)d_in[1];
    const float* norm_b   = (const float*)d_in[2];
    const float* in_pw    = (const float*)d_in[3];   // (256, 64)
    const float* conv_w   = (const float*)d_in[4];   // (128, 4)
    const float* conv_b   = (const float*)d_in[5];
    const float* x_pw     = (const float*)d_in[6];   // (36, 128)
    const float* dt_pw    = (const float*)d_in[7];   // (128, 4)
    const float* dt_pb    = (const float*)d_in[8];
    const float* A_log    = (const float*)d_in[9];   // (128, 16)
    const float* Dp       = (const float*)d_in[10];
    const float* out_pw   = (const float*)d_in[11];  // (64, 128)
    const float* proj_w   = (const float*)d_in[12];  // (256, 256)
    const float* proj_b   = (const float*)d_in[13];
    const float* skip     = (const float*)d_in[14];
    float* out = (float*)d_out;

    // workspace carve-up
    char* ws = (char*)d_ws;
    size_t off = 0;
    auto carve = [&](size_t bytes) { char* p = ws + off; off = (off + bytes + 255) & ~(size_t)255; return p; };
    float*          xflat    = (float*)carve((size_t)BATCH * NPOS * C_IN * 4);
    float*          chunks   = (float*)carve((size_t)MB * NPOS * DMODEL * 4);
    unsigned short* chunksbf = (unsigned short*)carve((size_t)MB * NPOS * DMODEL * 2);
    float*          xz       = (float*)carve((size_t)MB * NPOS * 256 * 4);
    float*          xi       = (float*)carve((size_t)MB * NPOS * DINNER * 4);
    float*          dt       = (float*)carve((size_t)MB * NPOS * DINNER * 4);
    float*          Bm       = (float*)carve((size_t)MB * NPOS * DSTATE * 4);
    float*          Cm       = (float*)carve((size_t)MB * NPOS * DSTATE * 4);
    float*          ysb      = (float*)carve((size_t)MB * NPOS * DINNER * 4);
    unsigned short* gbuf     = (unsigned short*)carve((size_t)MB * NPOS * DINNER * 2);
    float*          yo       = (float*)carve((size_t)MB * NPOS * DMODEL * 4);
    unsigned short* xmb      = (unsigned short*)carve((size_t)BATCH * NPOS * C_IN * 2);
    unsigned short* wbf      = (unsigned short*)carve((size_t)90112 * 2);
    unsigned short* in_pw_bf  = wbf;           // 256*64
    unsigned short* out_pw_bf = wbf + 16384;   // 64*128
    unsigned short* proj_w_bf = wbf + 24576;   // 256*256
    (void)ws_size; (void)n_in; (void)in_sizes; (void)out_size;

    // P: weights -> bf16 (tiny, one-shot per call)
    k_prep_w<<<(90112 + 255) / 256, 256, 0, stream>>>(in_pw, out_pw, proj_w, wbf);
    // 0: transpose to (B, n, C)
    k_transpose<<<dim3(NPOS / 32, C_IN / 32, BATCH), dim3(32, 8), 0, stream>>>(x, xflat);
    // 1: LN -> chunks (fp32 + bf16)
    k_ln1<<<BATCH * NPOS, 256, 0, stream>>>(xflat, norm_w, norm_b, chunks, chunksbf);
    // 2: in_proj  (M=4096, N=256, K=64) x 32 batches
    k_gemm_bf16<<<dim3(NPOS / 128, 256 / 16, MB), 256, 0, stream>>>(
        chunksbf, in_pw_bf, xz, /*K=*/DMODEL, /*lda=*/DMODEL, /*ldw=*/DMODEL, /*ldc=*/256,
        (long)NPOS * DMODEL, (long)NPOS * 256);
    // 3: conv + SiLU
    k_conv_silu<<<(MB * NPOS * DINNER) / 256, 256, 0, stream>>>(xz, conv_w, conv_b, xi);
    // 4: x_proj + dt_proj + softplus
    k_xproj_dt<<<MB * NPOS, DINNER, 0, stream>>>(xi, x_pw, dt_pw, dt_pb, dt, Bm, Cm);
    // 5: selective scan
    k_scan<<<MB, DINNER, 0, stream>>>(dt, Bm, Cm, xi, A_log, ysb);
    // 6: gate -> bf16
    k_gate<<<(MB * NPOS * DINNER) / 256, 256, 0, stream>>>(ysb, xi, xz, Dp, gbuf);
    // 7: out_proj (M=4096, N=64, K=128) x 32 batches
    k_gemm_bf16<<<dim3(NPOS / 128, DMODEL / 16, MB), 256, 0, stream>>>(
        gbuf, out_pw_bf, yo, /*K=*/DINNER, /*lda=*/DINNER, /*ldw=*/DINNER, /*ldc=*/DMODEL,
        (long)NPOS * DINNER, (long)NPOS * DMODEL);
    // 8: rotary + skip + recombine + LN -> xm (bf16)
    k_rot_skip_ln<<<BATCH * NPOS, 256, 0, stream>>>(yo, chunks, xflat, norm_w, norm_b, skip, xmb);
    // 9: final proj (M=32768, N=256, K=256) + bias, transposed store
    k_gemm_final<<<dim3((BATCH * NPOS) / 128, C_IN / 16, 1), 256, 0, stream>>>(
        xmb, proj_w_bf, proj_b, out);
}
